// ViSiLHead_31696858644516
// MI455X (gfx1250) — matrix operand order repaired and checked by reading, not compiled, not run on hardware
//
#include <hip/hip_runtime.h>

typedef __attribute__((ext_vector_type(16))) __bf16 v16bf;
typedef __attribute__((ext_vector_type(8)))  float  v8f;

#define DDIM  3840
#define NROWS 4608   // 512 * 9

#define ROWDW  20              // padded LDS row stride in dwords (80 B) -> conflict-free gathers
#define SLABDW (128 * ROWDW)   // 2560 dwords per slab (Ah / Al / Bh / Bl)
#define BUFDW  (4 * SLABDW)    // 10240 dwords per double-buffer half

// ---------------- helpers ----------------
__device__ __forceinline__ unsigned short f2bf_rne(float f) {
  unsigned int u = __float_as_uint(f);
  unsigned int r = (u + 0x7FFFu + ((u >> 16) & 1u)) >> 16;
  return (unsigned short)r;
}
__device__ __forceinline__ float bf2f(unsigned short h) {
  return __uint_as_float(((unsigned int)h) << 16);
}
// float atomic max via int/uint ordering trick (init value must be 0xFF800000 = -inf)
__device__ __forceinline__ void atomicMaxF(int* addr, float v) {
  int iv = __float_as_int(v);
  if (iv >= 0) atomicMax(addr, iv);
  else         atomicMin((unsigned int*)addr, (unsigned int)iv);
}
// CDNA5 async copy: global -> LDS, 16B per lane, tracked by ASYNCcnt
__device__ __forceinline__ void async_ld_b128(unsigned lds_byte_off, const void* gaddr) {
  asm volatile("global_load_async_to_lds_b128 %0, %1, off"
               :: "v"(lds_byte_off), "v"(gaddr) : "memory");
}
__device__ __forceinline__ void wait_async0() {
  asm volatile("s_wait_asynccnt 0" ::: "memory");
}

// ---------------- init rowmax to -inf ----------------
__global__ void init_rowmax(int* __restrict__ rm, int n) {
  int i = blockIdx.x * blockDim.x + threadIdx.x;
  if (i < n) rm[i] = 0xFF800000;
}

// ---------------- attention scale + bf16 hi/lo split ----------------
__global__ void __launch_bounds__(256) attn_split(
    const float* __restrict__ q, const float* __restrict__ t, const float* __restrict__ v,
    unsigned short* __restrict__ qh, unsigned short* __restrict__ ql,
    unsigned short* __restrict__ th, unsigned short* __restrict__ tl)
{
  __shared__ float red[256];
  int row = blockIdx.x;
  const float* src; unsigned short *dh, *dl;
  if (row < NROWS) { src = q + (size_t)row * DDIM; dh = qh + (size_t)row * DDIM; dl = ql + (size_t)row * DDIM; }
  else { int r2 = row - NROWS; src = t + (size_t)r2 * DDIM; dh = th + (size_t)r2 * DDIM; dl = tl + (size_t)r2 * DDIM; }
  int tid = threadIdx.x;
  float s = 0.f;
  for (int k = tid; k < DDIM; k += 256) s += src[k] * v[k];
  red[tid] = s; __syncthreads();
  for (int off = 128; off > 0; off >>= 1) {
    if (tid < off) red[tid] += red[tid + off];
    __syncthreads();
  }
  float w = red[0] * 0.5f + 0.5f;
  for (int k = tid; k < DDIM; k += 256) {
    float x = src[k] * w;
    unsigned short hi = f2bf_rne(x);
    unsigned short lo = f2bf_rne(x - bf2f(hi));
    dh[k] = hi; dl[k] = lo;
  }
}

// ---------------- WMMA GEMM (bf16x3), async-LDS double-buffered, fused max epilogue ----------------
union Frag { unsigned int u[8]; v16bf v; };

__global__ void __launch_bounds__(256) sim_gemm_max(
    const unsigned short* __restrict__ qh, const unsigned short* __restrict__ ql,
    const unsigned short* __restrict__ th, const unsigned short* __restrict__ tl,
    int* __restrict__ g_rowmax /* [NROWS][9] float bits */)
{
  __shared__ __align__(16) unsigned int smem[2 * BUFDW];  // 81,920 B staging
  __shared__ int runmax[128 * 9];
  const int tid = threadIdx.x;
  for (int i = tid; i < 1152; i += 256) runmax[i] = 0xFF800000;

  const int lane  = tid & 31;
  const int wv    = tid >> 5;
  const int mhalf = wv >> 2;      // 2 M halves of 64 rows
  const int nq    = wv & 3;       // 4 N quarters of 32 cols
  const int half  = lane >> 4;
  const int l16   = lane & 15;
  const int Mbase = blockIdx.x * 128;

  // dword offsets (within a 32-K chunk), per ISA 7.12.2 16-bit layouts
  int offA[8], offB[8];
#pragma unroll
  for (int r = 0; r < 8; ++r) {
    offA[r] = 8 * (r >> 2) + 4 * half + (r & 3);  // A: K = 16*(r/4) + 8*half + 2*(r%4) + slot
    offB[r] = 8 * (r >> 2) + 4 * half + (r & 3);  // B: K = 16*(r/4) + 8*half + 2*(r%4) + slot
  }
  int aRow[4], bRow[2];
#pragma unroll
  for (int mt = 0; mt < 4; ++mt) aRow[mt] = (mhalf * 64 + mt * 16 + l16) * ROWDW;
#pragma unroll
  for (int nt = 0; nt < 2; ++nt) bRow[nt] = (nq * 32 + nt * 16 + l16) * ROWDW;

  const unsigned sbase = (unsigned)(size_t)&smem[0];  // LDS byte offset (low 32 bits of aperture addr)
  const int srow = tid >> 2;   // staging: thread covers rows srow and srow+64, segment sseg
  const int sseg = tid & 3;

  for (int it = 0; it < 3; ++it) {
    const int NB = blockIdx.y * 384 + it * 128;

    // per-thread staging descriptors: 4 slabs x 2 row groups
    const unsigned short* gsrc[8];
    unsigned ldso[8];
#pragma unroll
    for (int i2 = 0; i2 < 2; ++i2) {
      const int row = srow + i2 * 64;
      gsrc[0 * 2 + i2] = qh + (size_t)(Mbase + row) * DDIM + sseg * 8;
      gsrc[1 * 2 + i2] = ql + (size_t)(Mbase + row) * DDIM + sseg * 8;
      gsrc[2 * 2 + i2] = th + (size_t)(NB    + row) * DDIM + sseg * 8;
      gsrc[3 * 2 + i2] = tl + (size_t)(NB    + row) * DDIM + sseg * 8;
#pragma unroll
      for (int s = 0; s < 4; ++s)
        ldso[s * 2 + i2] = sbase + 4u * (s * SLABDW + row * ROWDW + sseg * 4);
    }

    v8f acc[4][2];
#pragma unroll
    for (int mt = 0; mt < 4; ++mt)
#pragma unroll
      for (int nt = 0; nt < 2; ++nt)
        acc[mt][nt] = (v8f){0.f, 0.f, 0.f, 0.f, 0.f, 0.f, 0.f, 0.f};

    // prologue: stage K-step 0 into buffer 0
#pragma unroll
    for (int x = 0; x < 8; ++x)
      async_ld_b128(ldso[x], (const void*)gsrc[x]);
    wait_async0();
    __syncthreads();

    int cur = 0;
    for (int ks = 0; ks < 120; ++ks) {           // 120 K-steps of 32
      if (ks + 1 < 120) {                         // prefetch next step into other buffer
#pragma unroll
        for (int x = 0; x < 8; ++x)
          async_ld_b128(ldso[x] + (cur ^ 1) * (unsigned)(BUFDW * 4),
                        (const void*)((const char*)gsrc[x] + (size_t)(ks + 1) * 64));
      }
      const unsigned int* sb = smem + cur * BUFDW;
      Frag ah[4], al[4], bh[2], bl[2];
#pragma unroll
      for (int mt = 0; mt < 4; ++mt)
#pragma unroll
        for (int r = 0; r < 8; ++r) {
          ah[mt].u[r] = sb[0 * SLABDW + aRow[mt] + offA[r]];
          al[mt].u[r] = sb[1 * SLABDW + aRow[mt] + offA[r]];
        }
#pragma unroll
      for (int nt = 0; nt < 2; ++nt)
#pragma unroll
        for (int r = 0; r < 8; ++r) {
          bh[nt].u[r] = sb[2 * SLABDW + bRow[nt] + offB[r]];
          bl[nt].u[r] = sb[3 * SLABDW + bRow[nt] + offB[r]];
        }
#pragma unroll
      for (int mt = 0; mt < 4; ++mt)
#pragma unroll
        for (int nt = 0; nt < 2; ++nt) {
          acc[mt][nt] = __builtin_amdgcn_wmma_f32_16x16x32_bf16(false, ah[mt].v, false, bh[nt].v, (short)0, acc[mt][nt], false, false);
          acc[mt][nt] = __builtin_amdgcn_wmma_f32_16x16x32_bf16(false, ah[mt].v, false, bl[nt].v, (short)0, acc[mt][nt], false, false);
          acc[mt][nt] = __builtin_amdgcn_wmma_f32_16x16x32_bf16(false, al[mt].v, false, bh[nt].v, (short)0, acc[mt][nt], false, false);
        }
      wait_async0();
      __syncthreads();
      cur ^= 1;
    }

    // C layout: VGPR r holds M = r + 8*half, N = lane&15. Reduce max over j into (row, p=col%9).
#pragma unroll
    for (int mt = 0; mt < 4; ++mt) {
      const int lrow0 = mhalf * 64 + mt * 16 + 8 * half;
#pragma unroll
      for (int nt = 0; nt < 2; ++nt) {
        const int p = (NB + nq * 32 + nt * 16 + l16) % 9;
#pragma unroll
        for (int r = 0; r < 8; ++r)
          atomicMaxF(&runmax[(lrow0 + r) * 9 + p], acc[mt][nt][r]);
      }
    }
  }
  __syncthreads();
  for (int i = tid; i < 1152; i += 256)
    atomicMaxF(&g_rowmax[Mbase * 9 + i], __int_as_float(runmax[i]));
}

// ---------------- mean over o: S[i][p] = mean_o rowmax[i*9+o][p] ----------------
__global__ void mean_o(const int* __restrict__ rm, float* __restrict__ S) {
  int idx = blockIdx.x * blockDim.x + threadIdx.x;
  if (idx >= 4608) return;
  int i = idx / 9, p = idx % 9;
  float s = 0.f;
  for (int o = 0; o < 9; ++o) s += __int_as_float(rm[(i * 9 + o) * 9 + p]);
  S[idx] = s * (1.f / 9.f);
}

// ---------------- direct 3x3 conv (pad 1) + relu ----------------
__global__ void conv3x3_relu(const float* __restrict__ in, const float* __restrict__ w,
                             const float* __restrict__ b, float* __restrict__ out,
                             int Cin, int H, int W, int Cout) {
  int idx = blockIdx.x * blockDim.x + threadIdx.x;
  int total = Cout * H * W;
  if (idx >= total) return;
  int x = idx % W, h = (idx / W) % H, co = idx / (W * H);
  float acc = b[co];
  for (int ci = 0; ci < Cin; ++ci) {
    const float* ip = in + (size_t)ci * H * W;
    const float* wp = w + ((size_t)co * Cin + ci) * 9;
#pragma unroll
    for (int kh = 0; kh < 3; ++kh) {
      int ih = h + kh - 1;
      if (ih < 0 || ih >= H) continue;
#pragma unroll
      for (int kw = 0; kw < 3; ++kw) {
        int iw = x + kw - 1;
        if (iw < 0 || iw >= W) continue;
        acc += ip[ih * W + iw] * wp[kh * 3 + kw];
      }
    }
  }
  out[idx] = fmaxf(acc, 0.f);
}

// ---------------- 2x2 max pool (VALID) ----------------
__global__ void maxpool2k(const float* __restrict__ in, float* __restrict__ out, int C, int H, int W) {
  int Ho = H >> 1, Wo = W >> 1;
  int idx = blockIdx.x * blockDim.x + threadIdx.x;
  int total = C * Ho * Wo;
  if (idx >= total) return;
  int wo = idx % Wo, ho = (idx / Wo) % Ho, c = idx / (Wo * Ho);
  const float* base = in + ((size_t)c * H + 2 * ho) * W + 2 * wo;
  out[idx] = fmaxf(fmaxf(base[0], base[1]), fmaxf(base[W], base[W + 1]));
}

// ---------------- 1x1 conv + clip + max over W + mean over H -> scalar ----------------
__global__ void __launch_bounds__(256) convf_final(const float* __restrict__ x3,
    const float* __restrict__ wf, const float* __restrict__ bfv, float* __restrict__ out) {
  __shared__ float buf[256];
  __shared__ float red[128];
  int tid = threadIdx.x;           // 256 threads = 128 h * 2 w
  int h = tid >> 1, w = tid & 1;
  float acc = bfv[0];
  for (int c = 0; c < 128; ++c) acc += x3[((size_t)c * 128 + h) * 2 + w] * wf[c];
  acc = fminf(fmaxf(acc, -1.f), 1.f);
  buf[tid] = acc; __syncthreads();
  if (tid < 128) red[tid] = fmaxf(buf[2 * tid], buf[2 * tid + 1]);
  __syncthreads();
  for (int off = 64; off > 0; off >>= 1) {
    if (tid < off) red[tid] += red[tid + off];
    __syncthreads();
  }
  if (tid == 0) out[0] = red[0] * (1.f / 128.f);
}

// ---------------- launch ----------------
extern "C" void kernel_launch(void* const* d_in, const int* in_sizes, int n_in,
                              void* d_out, int out_size, void* d_ws, size_t ws_size,
                              hipStream_t stream) {
  const float* q  = (const float*)d_in[0];
  const float* t  = (const float*)d_in[1];
  const float* av = (const float*)d_in[2];
  const float* w1 = (const float*)d_in[3];
  const float* b1 = (const float*)d_in[4];
  const float* w2 = (const float*)d_in[5];
  const float* b2 = (const float*)d_in[6];
  const float* w3 = (const float*)d_in[7];
  const float* b3 = (const float*)d_in[8];
  const float* wf = (const float*)d_in[9];
  const float* bf = (const float*)d_in[10];
  float* out = (float*)d_out;

  char* ws = (char*)d_ws;
  const size_t HALF = (size_t)NROWS * DDIM * sizeof(unsigned short); // 35,389,440 B
  unsigned short* qh  = (unsigned short*)(ws);
  unsigned short* ql  = (unsigned short*)(ws + HALF);
  unsigned short* th_ = (unsigned short*)(ws + 2 * HALF);
  unsigned short* tl_ = (unsigned short*)(ws + 3 * HALF);
  size_t off = 4 * HALF;
  int*   rm = (int*)(ws + off);            off += (size_t)NROWS * 9 * sizeof(int);     // 165,888
  float* S  = (float*)(ws + off);          off += (size_t)512 * 9 * sizeof(float);     // 18,432
  float* x1 = (float*)(ws + off);          off += (size_t)32 * 512 * 9 * sizeof(float);
  float* p1 = (float*)(ws + off);          off += (size_t)32 * 256 * 4 * sizeof(float);
  float* x2 = (float*)(ws + off);          off += (size_t)64 * 256 * 4 * sizeof(float);
  float* p2 = (float*)(ws + off);          off += (size_t)64 * 128 * 2 * sizeof(float);
  float* x3 = (float*)(ws + off);          off += (size_t)128 * 128 * 2 * sizeof(float);

  init_rowmax<<<(NROWS * 9 + 255) / 256, 256, 0, stream>>>(rm, NROWS * 9);
  attn_split<<<2 * NROWS, 256, 0, stream>>>(q, t, av, qh, ql, th_, tl_);
  sim_gemm_max<<<dim3(36, 12), 256, 0, stream>>>(qh, ql, th_, tl_, rm);
  mean_o<<<(4608 + 255) / 256, 256, 0, stream>>>(rm, S);
  conv3x3_relu<<<(32 * 512 * 9 + 255) / 256, 256, 0, stream>>>(S, w1, b1, x1, 1, 512, 9, 32);
  maxpool2k<<<(32 * 256 * 4 + 255) / 256, 256, 0, stream>>>(x1, p1, 32, 512, 9);
  conv3x3_relu<<<(64 * 256 * 4 + 255) / 256, 256, 0, stream>>>(p1, w2, b2, x2, 32, 256, 4, 64);
  maxpool2k<<<(64 * 128 * 2 + 255) / 256, 256, 0, stream>>>(x2, p2, 64, 256, 4);
  conv3x3_relu<<<(128 * 128 * 2 + 255) / 256, 256, 0, stream>>>(p2, w3, b3, x3, 64, 128, 2, 128);
  convf_final<<<1, 256, 0, stream>>>(x3, wf, bf, out);
}
